// HarmonicOscillator1D_72885595013422
// MI455X (gfx1250) — matrix operand order repaired and checked by reading, not compiled, not run on hardware
//
#include <hip/hip_runtime.h>
#include <hip/hip_bf16.h>

// ---------- types ----------
typedef __attribute__((ext_vector_type(16))) _Float16 v16h;
typedef __attribute__((ext_vector_type(8)))  _Float16 v8h;
typedef __attribute__((ext_vector_type(8)))  float    v8f;

__device__ __forceinline__ v16h cat16(v8h a, v8h b) {
    return __builtin_shufflevector(a, b, 0,1,2,3,4,5,6,7,8,9,10,11,12,13,14,15);
}

// D = A(16x32 f16) * B(32x16 f16) + C(16x16 f32)   [probe-confirmed builtin]
__device__ __forceinline__ v8f wmma_f16(v16h a, v16h b, v8f c) {
    return __builtin_amdgcn_wmma_f32_16x16x32_f16(false, a, false, b, (short)0, c,
                                                  false, false);
}

// physics constants (N_INPUT=8 -> hbar*omega = 40*8^(-1/3) = 20 exactly)
#define KINC (-(197.0f*197.0f)/(2.0f*939.0f))
#define POTC (0.5f*939.0f*(20.0f/197.0f)*(20.0f/197.0f))

// ---------------------------------------------------------------------------
// Prep kernel: W2 (f32 [256][256] row-major, rows = K dim for H1@W2)
//   w2h : f16 row-major copy (used as WMMA *A* operand in V1 = W2 @ U2)
//   w2b : f16 pre-swizzled into the wave32 B-operand register tiling:
//         tile (kt 0..7, nt 0..15); per lane l, dword v (0..7) packs
//         ( W2[kt*32 + 2v + 8*(l/16) + 8*(v/4)][nt*16 + l%16],
//           W2[kt*32 + 2v+1+ 8*(l/16) + 8*(v/4)][nt*16 + l%16] )
//         so a B tile load is one contiguous 32B read per lane.
// ---------------------------------------------------------------------------
__global__ void ho_prep_kernel(const float* __restrict__ W2,
                               _Float16* __restrict__ w2b,
                               _Float16* __restrict__ w2h) {
    int t = blockIdx.x * blockDim.x + threadIdx.x;   // 0..65535
    w2h[t] = (_Float16)W2[t];
    if (t < 32768) {                                  // one dword of w2b each
        int v  = t & 7;
        int l  = (t >> 3) & 31;
        int nt = (t >> 8) & 15;
        int kt = t >> 12;
        int k  = kt * 32 + 2 * v + ((l >> 4) << 3) + ((v >> 2) << 3);
        int n  = nt * 16 + (l & 15);
        w2b[2 * t    ] = (_Float16)W2[k * 256 + n];
        w2b[2 * t + 1] = (_Float16)W2[(k + 1) * 256 + n];
    }
}

// ---------------------------------------------------------------------------
// Main kernel: one 128-thread block handles 16 samples.
// ---------------------------------------------------------------------------
__global__ __launch_bounds__(128)
void ho_energy_kernel(const float* __restrict__ x,
                      const float* __restrict__ W1g,
                      const float* __restrict__ b1,
                      const float* __restrict__ b2,
                      const float* __restrict__ W3,
                      const _Float16* __restrict__ w2b,
                      const _Float16* __restrict__ w2h,
                      float* __restrict__ out) {
    // ---- LDS (≈63 KB) ----
    __shared__ alignas(32) _Float16 H1h[16][256];   // tanh(z1), f16 (WMMA A)
    __shared__ alignas(32) _Float16 D1h[16][256];   // 1 - h1^2
    __shared__ alignas(32) _Float16 W1h[8][256];    // W1 as f16
    __shared__ alignas(32) _Float16 U2h[16][256];   // (1-h2^2)*w3  (WMMA B)
    __shared__             _Float16 WL2h[16][256];  // h2*(1-h2^2)*w3
    __shared__             float    U1f[16][256];   // (1-h1^2)*v1
    __shared__             float    W1s[8][256];    // W1 f32
    __shared__             float    Ssum[256];      // sum_i W1[i][j]^2
    __shared__             float    w3s[256];
    __shared__             float    Xs[16][8];
    __shared__             float    xsq[16], lap1acc[16], lap2acc[16], gsqacc[16];

    const int t    = threadIdx.x;
    const int lane = t & 31;
    const int wv   = t >> 5;          // wave id 0..3
    const int g    = blockIdx.x;      // sample group (16 samples)

    // ---- stage 0: load params / init ----
    Xs[t >> 3][t & 7] = x[g * 128 + t];
    for (int idx = t; idx < 2048; idx += 128) {
        float w = W1g[idx];
        (&W1s[0][0])[idx] = w;
        (&W1h[0][0])[idx] = (_Float16)w;
    }
    for (int idx = t; idx < 256; idx += 128) w3s[idx] = W3[idx];
    if (t < 16) { lap1acc[t] = 0.f; lap2acc[t] = 0.f; gsqacc[t] = 0.f; }
    __syncthreads();
    for (int j = t; j < 256; j += 128) {
        float s = 0.f;
        for (int i = 0; i < 8; ++i) { float w = W1s[i][j]; s += w * w; }
        Ssum[j] = s;
    }
    if (t < 16) {
        float q = 0.f;
        for (int i = 0; i < 8; ++i) { float xi = Xs[t][i]; q += xi * xi; }
        xsq[t] = q;
    }
    __syncthreads();

    // ---- stage 1: z1 = x@W1+b1 (K=8, VALU), h1 = tanh ----
    for (int idx = t; idx < 4096; idx += 128) {
        int s = idx >> 8, n = idx & 255;
        float z = b1[n];
        for (int i = 0; i < 8; ++i) z += Xs[s][i] * W1s[i][n];
        float h = tanhf(z);
        H1h[s][n] = (_Float16)h;
        D1h[s][n] = (_Float16)(1.f - h * h);
    }
    __syncthreads();

    // ---- stage 2: Z2 = H1 @ W2 (WMMA), h2/u2/wl2 ----
    {
        const int srow = lane & 15;              // A row = sample
        const int o    = (lane >> 4) << 3;       // K-run offset per half-wave
        for (int nt = wv; nt < 16; nt += 4) {
            v8f acc = {};
            for (int kt = 0; kt < 8; ++kt) {
                int k0 = kt * 32;
                v8h alo = *(const v8h*)&H1h[srow][k0 + o];
                v8h ahi = *(const v8h*)&H1h[srow][k0 + 16 + o];
                v16h b  = *(const v16h*)(w2b + (((kt * 16 + nt) * 32 + lane) << 4));
                acc = wmma_f16(cat16(alo, ahi), b, acc);
            }
            int n = nt * 16 + (lane & 15);
            float b2n = b2[n], w3n = w3s[n];
            for (int v = 0; v < 8; ++v) {
                int s  = v + ((lane >> 4) << 3);
                float h = tanhf(acc[v] + b2n);
                float d = 1.f - h * h;
                U2h[s][n]  = (_Float16)(d * w3n);
                WL2h[s][n] = (_Float16)(h * d * w3n);
            }
        }
    }
    __syncthreads();

    // ---- stage 3: V1 = W2 @ U2 (WMMA, A = W2 rows), u1, lap term 1 ----
    {
        float lp1 = 0.f;
        const int o  = (lane >> 4) << 4;
        const int s  = lane & 15;
        for (int jt = wv; jt < 16; jt += 4) {
            v8f acc = {};
            const int jr = jt * 16 + (lane & 15);          // A row
            for (int kt = 0; kt < 8; ++kt) {
                int k0 = kt * 32;
                v8h alo = *(const v8h*)(w2h + jr * 256 + k0 + o);
                v8h ahi = *(const v8h*)(w2h + jr * 256 + k0 + 8 + o);
                v16h b  = *(const v16h*)&U2h[s][k0 + ((lane >> 4) << 4)];
                acc = wmma_f16(cat16(alo, ahi), b, acc);
            }
            for (int v = 0; v < 8; ++v) {
                int j = jt * 16 + v + ((lane >> 4) << 3);
                float v1 = acc[v];
                float d1 = (float)D1h[s][j];
                float h1 = (float)H1h[s][j];
                U1f[s][j] = d1 * v1;
                lp1 += Ssum[j] * h1 * d1 * v1;
            }
        }
        lp1 += __shfl_xor(lp1, 16);                 // lanes l, l+16 share sample
        if (lane < 16) atomicAdd(&lap1acc[lane], lp1);
    }
    __syncthreads();

    // ---- stage 4: g = W1 @ u1, |g|^2 ----
    {
        int i = t >> 4, s = t & 15;                 // exactly 128 (i,s) pairs
        float gi = 0.f;
        for (int j = 0; j < 256; ++j) gi += W1s[i][j] * U1f[s][j];
        atomicAdd(&gsqacc[s], gi * gi);
    }

    // ---- stage 5: B = (D1 .* W1row) @ W2, lap term 2 (never materialized) ----
    {
        const int M  = lane & 15;
        const int o  = (lane >> 4) << 3;
        const int nc = lane & 15;
        for (int m = wv; m < 8; m += 4) {           // 2 samples per tile
            const int sA = 2 * m + (M >> 3);        // A-row sample
            const int iA = M & 7;                   // A-row input dim
            const int sC = 2 * m + (lane >> 4);     // C-layout sample of this lane
            float lp2 = 0.f;
            for (int nt = 0; nt < 16; ++nt) {
                v8f acc = {};
                for (int kt = 0; kt < 8; ++kt) {
                    int k0 = kt * 32;
                    v8h a0 = *(const v8h*)&D1h[sA][k0 + o]      * *(const v8h*)&W1h[iA][k0 + o];
                    v8h a1 = *(const v8h*)&D1h[sA][k0 + 16 + o] * *(const v8h*)&W1h[iA][k0 + 16 + o];
                    v16h b = *(const v16h*)(w2b + (((kt * 16 + nt) * 32 + lane) << 4));
                    acc = wmma_f16(cat16(a0, a1), b, acc);
                }
                float wl = (float)WL2h[sC][nt * 16 + nc];
                float q  = 0.f;
                for (int v = 0; v < 8; ++v) q += acc[v] * acc[v];  // sums over i
                lp2 += wl * q;
            }
            lp2 += __shfl_xor(lp2, 1);
            lp2 += __shfl_xor(lp2, 2);
            lp2 += __shfl_xor(lp2, 4);
            lp2 += __shfl_xor(lp2, 8);              // reduce over N within half
            if ((lane & 15) == 0) atomicAdd(&lap2acc[sC], lp2);
        }
    }
    __syncthreads();

    // ---- final: ek + ep ----
    if (t < 16) {
        float lap = -2.f * (lap1acc[t] + lap2acc[t]);
        float ek  = KINC * (lap + gsqacc[t]);
        out[g * 16 + t] = ek + POTC * xsq[t];
    }
}

// ---------------------------------------------------------------------------
extern "C" void kernel_launch(void* const* d_in, const int* in_sizes, int n_in,
                              void* d_out, int out_size, void* d_ws, size_t ws_size,
                              hipStream_t stream) {
    const float* x  = (const float*)d_in[0];
    const float* W1 = (const float*)d_in[1];
    const float* b1 = (const float*)d_in[2];
    const float* W2 = (const float*)d_in[3];
    const float* b2 = (const float*)d_in[4];
    const float* W3 = (const float*)d_in[5];
    // b3 (d_in[6]) only shifts log|psi|; it never appears in derivatives or output.
    float* out = (float*)d_out;

    _Float16* w2b = (_Float16*)d_ws;          // 128 KB swizzled B tiles
    _Float16* w2h = w2b + 65536;              // 128 KB f16 row-major W2

    ho_prep_kernel<<<256, 256, 0, stream>>>(W2, w2b, w2h);
    ho_energy_kernel<<<2048, 128, 0, stream>>>(x, W1, b1, b2, W3, w2b, w2h, out);
}
